// CausalMultiHeadSelfAttention_88691074662418
// MI455X (gfx1250) — matrix operand order repaired and checked by reading, not compiled, not run on hardware
//
#include <hip/hip_runtime.h>
#include <stdint.h>

// Causal MHA, B=4 S=2048 D=1024 H=16 dk=64, fp32 in/out, bf16 WMMA internally.
// Pipeline:
//   [cvt f32->bf16: x, WQ, WK, WV, WO]            (bandwidth-trivial pre-pass)
//   [qkv projections -> bf16 Q/K/Vt]              (bf16 WMMA GEMM, 32x64 wave tile)
//   [fused causal flash attention -> bf16 attn]   (WMMA + online softmax)
//   [WO projection -> f32 out]                    (bf16 WMMA GEMM, 32x64 wave tile)

#define BATCH 4
#define SEQ   2048
#define DMODEL 1024
#define HEADS 16
#define DK    64
#define BS    (BATCH * SEQ)   // 8192

typedef __attribute__((ext_vector_type(16))) __bf16 v16bf;
typedef __attribute__((ext_vector_type(8)))  __bf16 v8bf;
typedef __attribute__((ext_vector_type(8)))  float  v8f;

union FragBF {
    v16bf    v;
    uint4    q[2];
    uint32_t u[8];
};

union Pack8BF {
    v8bf  v;
    uint4 q;
};

__device__ __forceinline__ v8f wmma_bf16(const FragBF& a, const FragBF& b, v8f c) {
    return __builtin_amdgcn_wmma_f32_16x16x32_bf16(false, a.v, false, b.v,
                                                   (short)0, c, false, false);
}

// ---------------------------------------------------------------------------
// Kernel 0: f32 -> bf16 conversion pre-pass (8 elements / thread).
// ---------------------------------------------------------------------------
__global__ __launch_bounds__(256) void cvt_f32_bf16_kernel(
    const float* __restrict__ src, __bf16* __restrict__ dst)
{
    const size_t i = ((size_t)blockIdx.x * blockDim.x + threadIdx.x) * 8;
    float4 f0 = *(const float4*)(src + i);
    float4 f1 = *(const float4*)(src + i + 4);
    Pack8BF p;
    p.v[0] = (__bf16)f0.x; p.v[1] = (__bf16)f0.y;
    p.v[2] = (__bf16)f0.z; p.v[3] = (__bf16)f0.w;
    p.v[4] = (__bf16)f1.x; p.v[5] = (__bf16)f1.y;
    p.v[6] = (__bf16)f1.z; p.v[7] = (__bf16)f1.w;
    *(uint4*)(dst + i) = p.q;
}

// ---------------------------------------------------------------------------
// Kernel 1: Q/K/V projections.  y = x @ W^T  (W is [out,in]), all-bf16 inputs.
// Block = 256 thr (8 waves); block tile 64(rows) x 256(cols);
// wave tile 32x64 = 2 M-tiles x 4 N-tiles (8 WMMA accumulators).
// blockIdx.z selects {Q,K,V}.  Q,K stored bf16 [b,h,s,64]; V stored bf16 [b,h,64,s].
// ---------------------------------------------------------------------------
__global__ __launch_bounds__(256) void qkv_proj_kernel(
    const __bf16* __restrict__ xb,
    const __bf16* __restrict__ WQb, const __bf16* __restrict__ WKb,
    const __bf16* __restrict__ WVb,
    __bf16* __restrict__ Qb, __bf16* __restrict__ Kb, __bf16* __restrict__ Vt)
{
    const int wave = threadIdx.x >> 5;
    const int lane = threadIdx.x & 31;
    const int half = lane >> 4;
    const int l16  = lane & 15;
    const int which = blockIdx.z;                       // 0=Q 1=K 2=V
    const __bf16* W = (which == 0) ? WQb : (which == 1) ? WKb : WVb;

    const int rowbase = blockIdx.x * 64 + (wave & 1) * 32;   // over B*S (2 m-tiles)
    const int colbase = blockIdx.y * 256 + (wave >> 1) * 64; // over D   (4 n-tiles)

    const __bf16* xrow0 = xb + (size_t)(rowbase + l16) * DMODEL;
    const __bf16* xrow1 = xrow0 + (size_t)16 * DMODEL;
    const __bf16* wr0 = W + (size_t)(colbase + l16) * DMODEL;
    const __bf16* wr1 = wr0 + (size_t)16 * DMODEL;
    const __bf16* wr2 = wr0 + (size_t)32 * DMODEL;
    const __bf16* wr3 = wr0 + (size_t)48 * DMODEL;

    v8f acc[2][4];
    #pragma unroll
    for (int i = 0; i < 2; ++i)
        #pragma unroll
        for (int j = 0; j < 4; ++j) acc[i][j] = (v8f){};

    for (int k0 = 0; k0 < DMODEL; k0 += 32) {
        FragBF a0, a1, b0, b1, b2, b3;
        // A 16x32 bf16: lane half h: j -> K = k0 + h*16 + j (contiguous in x row)
        a0.q[0] = *(const uint4*)(xrow0 + k0 + half * 16);
        a0.q[1] = *(const uint4*)(xrow0 + k0 + half * 16 + 8);
        a1.q[0] = *(const uint4*)(xrow1 + k0 + half * 16);
        a1.q[1] = *(const uint4*)(xrow1 + k0 + half * 16 + 8);
        // B 32x16 bf16: lane half h: j -> K = k0 + h*16 + j (contiguous in W row)
        b0.q[0] = *(const uint4*)(wr0 + k0 + half * 16);
        b0.q[1] = *(const uint4*)(wr0 + k0 + half * 16 + 8);
        b1.q[0] = *(const uint4*)(wr1 + k0 + half * 16);
        b1.q[1] = *(const uint4*)(wr1 + k0 + half * 16 + 8);
        b2.q[0] = *(const uint4*)(wr2 + k0 + half * 16);
        b2.q[1] = *(const uint4*)(wr2 + k0 + half * 16 + 8);
        b3.q[0] = *(const uint4*)(wr3 + k0 + half * 16);
        b3.q[1] = *(const uint4*)(wr3 + k0 + half * 16 + 8);

        acc[0][0] = wmma_bf16(a0, b0, acc[0][0]);
        acc[0][1] = wmma_bf16(a0, b1, acc[0][1]);
        acc[0][2] = wmma_bf16(a0, b2, acc[0][2]);
        acc[0][3] = wmma_bf16(a0, b3, acc[0][3]);
        acc[1][0] = wmma_bf16(a1, b0, acc[1][0]);
        acc[1][1] = wmma_bf16(a1, b1, acc[1][1]);
        acc[1][2] = wmma_bf16(a1, b2, acc[1][2]);
        acc[1][3] = wmma_bf16(a1, b3, acc[1][3]);
    }

    // C layout per tile: lane holds col n, rows (half*8 + r), r = 0..7
    #pragma unroll
    for (int i = 0; i < 2; ++i) {
        const int grow = rowbase + i * 16 + half * 8;
        const int bidx = grow >> 11;          // / SEQ
        const int s    = grow & (SEQ - 1);
        #pragma unroll
        for (int j = 0; j < 4; ++j) {
            const int n = colbase + j * 16 + l16;
            const int h = n >> 6;             // head
            const int d = n & 63;             // dim within head
            if (which == 2) {
                Pack8BF p;
                #pragma unroll
                for (int r = 0; r < 8; ++r) p.v[r] = (__bf16)acc[i][j][r];
                *(uint4*)(Vt + ((((size_t)bidx * HEADS + h) * DK + d) * SEQ + s)) = p.q;
            } else {
                __bf16* dstb = (which == 0) ? Qb : Kb;
                const size_t base = ((size_t)bidx * HEADS + h) * SEQ;
                #pragma unroll
                for (int r = 0; r < 8; ++r)
                    dstb[(base + s + r) * DK + d] = (__bf16)acc[i][j][r];
            }
        }
    }
}

// ---------------------------------------------------------------------------
// Kernel 2: fused causal flash attention.
// Block = 128 thr (4 waves); each wave owns one 16-row q tile; block covers 64
// q rows of one (b,h).  Key blocks of 32; online softmax; P transposed via LDS.
// Output: attnb bf16 [b, s, h*64+d]  (merged-head layout, ready for WO GEMM).
// ---------------------------------------------------------------------------
__global__ __launch_bounds__(128) void flash_attn_kernel(
    const __bf16* __restrict__ Qb,
    const __bf16* __restrict__ Kb,
    const __bf16* __restrict__ Vt,
    __bf16* __restrict__ attnb)
{
    __shared__ __bf16 lds_p[4][16 * 32];   // per-wave P tile (bf16)

    const int wave = threadIdx.x >> 5;
    const int lane = threadIdx.x & 31;
    const int half = lane >> 4;
    const int l16  = lane & 15;
    const int b = blockIdx.z, h = blockIdx.y;
    const int qbase = blockIdx.x * 64 + wave * 16;

    const size_t bh = (size_t)b * HEADS + h;
    const __bf16* Qh = Qb + bh * SEQ * DK;
    const __bf16* Kh = Kb + bh * SEQ * DK;
    const __bf16* Vh = Vt + bh * DK * SEQ;

    // Q A-fragments for dk 0..31 and 32..63 (held in registers for whole loop)
    FragBF aq0, aq1;
    {
        const __bf16* qrow = Qh + (size_t)(qbase + l16) * DK;
        aq0.q[0] = *(const uint4*)(qrow + half * 16);
        aq0.q[1] = *(const uint4*)(qrow + half * 16 + 8);
        aq1.q[0] = *(const uint4*)(qrow + 32 + half * 16);
        aq1.q[1] = *(const uint4*)(qrow + 40 + half * 16);
    }

    v8f o0 = {}, o1 = {}, o2 = {}, o3 = {};      // 16 x 64 f32 accumulator
    float mrow[8], lrow[8];
    #pragma unroll
    for (int r = 0; r < 8; ++r) { mrow[r] = -3.0e38f; lrow[r] = 0.0f; }
    const float scale = 0.125f;                  // 1/sqrt(64)
    const int q0 = qbase + half * 8;             // first C-row this half owns

    for (int kb = 0; kb <= qbase + 15; kb += 32) {
        // ---- scores: S = Q (16x64) x K^T (64x32) as two 16x16 C tiles ----
        v8f s0 = {}, s1 = {};
        {
            FragBF bk;
            const __bf16* krow0 = Kh + (size_t)(kb + l16) * DK + half * 16;
            bk.q[0] = *(const uint4*)(krow0);
            bk.q[1] = *(const uint4*)(krow0 + 8);
            s0 = wmma_bf16(aq0, bk, s0);
            bk.q[0] = *(const uint4*)(krow0 + 32);
            bk.q[1] = *(const uint4*)(krow0 + 40);
            s0 = wmma_bf16(aq1, bk, s0);

            const __bf16* krow1 = Kh + (size_t)(kb + 16 + l16) * DK + half * 16;
            bk.q[0] = *(const uint4*)(krow1);
            bk.q[1] = *(const uint4*)(krow1 + 8);
            s1 = wmma_bf16(aq0, bk, s1);
            bk.q[0] = *(const uint4*)(krow1 + 32);
            bk.q[1] = *(const uint4*)(krow1 + 40);
            s1 = wmma_bf16(aq1, bk, s1);
        }

        // ---- scale + causal mask (only on partially-visible blocks) ----
        if (kb + 31 > qbase) {
            const int k0c = kb + l16, k1c = kb + 16 + l16;
            #pragma unroll
            for (int r = 0; r < 8; ++r) {
                const int q = q0 + r;
                s0[r] = (k0c <= q) ? s0[r] * scale : -1e9f;
                s1[r] = (k1c <= q) ? s1[r] * scale : -1e9f;
            }
        } else {
            #pragma unroll
            for (int r = 0; r < 8; ++r) { s0[r] *= scale; s1[r] *= scale; }
        }

        // ---- online softmax; stash P (bf16) into per-wave LDS tile ----
        __bf16* lp = &lds_p[wave][0];
        #pragma unroll
        for (int r = 0; r < 8; ++r) {
            float mx = fmaxf(s0[r], s1[r]);
            mx = fmaxf(mx, __shfl_xor(mx, 1));
            mx = fmaxf(mx, __shfl_xor(mx, 2));
            mx = fmaxf(mx, __shfl_xor(mx, 4));
            mx = fmaxf(mx, __shfl_xor(mx, 8));
            const float mnew  = fmaxf(mrow[r], mx);
            const float p0    = __expf(s0[r] - mnew);
            const float p1    = __expf(s1[r] - mnew);
            const float alpha = __expf(mrow[r] - mnew);
            float ps = p0 + p1;
            ps += __shfl_xor(ps, 1);
            ps += __shfl_xor(ps, 2);
            ps += __shfl_xor(ps, 4);
            ps += __shfl_xor(ps, 8);
            lrow[r] = lrow[r] * alpha + ps;
            mrow[r] = mnew;
            o0[r] *= alpha; o1[r] *= alpha; o2[r] *= alpha; o3[r] *= alpha;
            lp[(half * 8 + r) * 32 + l16]      = (__bf16)p0;
            lp[(half * 8 + r) * 32 + 16 + l16] = (__bf16)p1;
        }
        asm volatile("s_wait_dscnt 0" ::: "memory");  // LDS in-order per wave

        // ---- re-read P as A-fragment (transpose through LDS) ----
        FragBF ap;
        {
            const __bf16* rp = lp + l16 * 32 + half * 16;
            ap.q[0] = *(const uint4*)(rp);
            ap.q[1] = *(const uint4*)(rp + 8);
        }

        // ---- O += P (16x32) x V (32x64): Vt gives contiguous B-fragments ----
        {
            FragBF bv;
            const __bf16* vcol = Vh + (size_t)l16 * SEQ + kb + half * 16;
            bv.q[0] = *(const uint4*)(vcol);
            bv.q[1] = *(const uint4*)(vcol + 8);
            o0 = wmma_bf16(ap, bv, o0);
            bv.q[0] = *(const uint4*)(vcol + 16 * SEQ);
            bv.q[1] = *(const uint4*)(vcol + 16 * SEQ + 8);
            o1 = wmma_bf16(ap, bv, o1);
            bv.q[0] = *(const uint4*)(vcol + 32 * SEQ);
            bv.q[1] = *(const uint4*)(vcol + 32 * SEQ + 8);
            o2 = wmma_bf16(ap, bv, o2);
            bv.q[0] = *(const uint4*)(vcol + 48 * SEQ);
            bv.q[1] = *(const uint4*)(vcol + 48 * SEQ + 8);
            o3 = wmma_bf16(ap, bv, o3);
        }
    }

    // ---- epilogue: normalize and write bf16 merged-head output ----
    const size_t colbase = (size_t)h * DK + l16;
    #pragma unroll
    for (int r = 0; r < 8; ++r) {
        const float inv = 1.0f / lrow[r];
        const size_t rowoff = ((size_t)b * SEQ + q0 + r) * DMODEL;
        attnb[rowoff + colbase +  0] = (__bf16)(o0[r] * inv);
        attnb[rowoff + colbase + 16] = (__bf16)(o1[r] * inv);
        attnb[rowoff + colbase + 32] = (__bf16)(o2[r] * inv);
        attnb[rowoff + colbase + 48] = (__bf16)(o3[r] * inv);
    }
}

// ---------------------------------------------------------------------------
// Kernel 3: output projection  out = attn @ WO^T  (bf16 A/B -> f32 out)
// Same 32x64 wave tile as kernel 1.
// ---------------------------------------------------------------------------
__global__ __launch_bounds__(256) void out_proj_kernel(
    const __bf16* __restrict__ attnb,
    const __bf16* __restrict__ WOb,
    float* __restrict__ out)
{
    const int wave = threadIdx.x >> 5;
    const int lane = threadIdx.x & 31;
    const int half = lane >> 4;
    const int l16  = lane & 15;

    const int rowbase = blockIdx.x * 64 + (wave & 1) * 32;
    const int colbase = blockIdx.y * 256 + (wave >> 1) * 64;

    const __bf16* ar0 = attnb + (size_t)(rowbase + l16) * DMODEL;
    const __bf16* ar1 = ar0 + (size_t)16 * DMODEL;
    const __bf16* wr0 = WOb + (size_t)(colbase + l16) * DMODEL;
    const __bf16* wr1 = wr0 + (size_t)16 * DMODEL;
    const __bf16* wr2 = wr0 + (size_t)32 * DMODEL;
    const __bf16* wr3 = wr0 + (size_t)48 * DMODEL;

    v8f acc[2][4];
    #pragma unroll
    for (int i = 0; i < 2; ++i)
        #pragma unroll
        for (int j = 0; j < 4; ++j) acc[i][j] = (v8f){};

    for (int k0 = 0; k0 < DMODEL; k0 += 32) {
        FragBF a0, a1, b0, b1, b2, b3;
        a0.q[0] = *(const uint4*)(ar0 + k0 + half * 16);
        a0.q[1] = *(const uint4*)(ar0 + k0 + half * 16 + 8);
        a1.q[0] = *(const uint4*)(ar1 + k0 + half * 16);
        a1.q[1] = *(const uint4*)(ar1 + k0 + half * 16 + 8);
        b0.q[0] = *(const uint4*)(wr0 + k0 + half * 16);
        b0.q[1] = *(const uint4*)(wr0 + k0 + half * 16 + 8);
        b1.q[0] = *(const uint4*)(wr1 + k0 + half * 16);
        b1.q[1] = *(const uint4*)(wr1 + k0 + half * 16 + 8);
        b2.q[0] = *(const uint4*)(wr2 + k0 + half * 16);
        b2.q[1] = *(const uint4*)(wr2 + k0 + half * 16 + 8);
        b3.q[0] = *(const uint4*)(wr3 + k0 + half * 16);
        b3.q[1] = *(const uint4*)(wr3 + k0 + half * 16 + 8);

        acc[0][0] = wmma_bf16(a0, b0, acc[0][0]);
        acc[0][1] = wmma_bf16(a0, b1, acc[0][1]);
        acc[0][2] = wmma_bf16(a0, b2, acc[0][2]);
        acc[0][3] = wmma_bf16(a0, b3, acc[0][3]);
        acc[1][0] = wmma_bf16(a1, b0, acc[1][0]);
        acc[1][1] = wmma_bf16(a1, b1, acc[1][1]);
        acc[1][2] = wmma_bf16(a1, b2, acc[1][2]);
        acc[1][3] = wmma_bf16(a1, b3, acc[1][3]);
    }

    #pragma unroll
    for (int i = 0; i < 2; ++i) {
        const int grow = rowbase + i * 16 + half * 8;
        #pragma unroll
        for (int j = 0; j < 4; ++j) {
            const int col = colbase + j * 16 + l16;
            #pragma unroll
            for (int r = 0; r < 8; ++r)
                out[(size_t)(grow + r) * DMODEL + col] = acc[i][j][r];
        }
    }
}

// ---------------------------------------------------------------------------
extern "C" void kernel_launch(void* const* d_in, const int* in_sizes, int n_in,
                              void* d_out, int out_size, void* d_ws, size_t ws_size,
                              hipStream_t stream) {
    const float* x  = (const float*)d_in[0];
    const float* WQ = (const float*)d_in[1];
    const float* WK = (const float*)d_in[2];
    const float* WV = (const float*)d_in[3];
    const float* WO = (const float*)d_in[4];
    float* out = (float*)d_out;

    // Workspace layout (bf16 elements):
    //   Q | K | Vt | attn | xb | WQb | WKb | WVb | WOb
    const size_t NQKV = (size_t)BATCH * HEADS * SEQ * DK;   // 8,388,608
    const size_t NX   = (size_t)BS * DMODEL;                // 8,388,608
    const size_t NW   = (size_t)DMODEL * DMODEL;            // 1,048,576
    __bf16* Qb    = (__bf16*)d_ws;
    __bf16* Kb    = Qb + NQKV;
    __bf16* Vt    = Kb + NQKV;
    __bf16* attnb = Vt + NQKV;
    __bf16* xb    = attnb + NX;
    __bf16* WQb   = xb + NX;
    __bf16* WKb   = WQb + NW;
    __bf16* WVb   = WKb + NW;
    __bf16* WOb   = WVb + NW;

    // 0) f32 -> bf16 conversion (8 elems / thread, 256 thr / block)
    cvt_f32_bf16_kernel<<<NX / (8 * 256), 256, 0, stream>>>(x, xb);
    cvt_f32_bf16_kernel<<<NW / (8 * 256), 256, 0, stream>>>(WQ, WQb);
    cvt_f32_bf16_kernel<<<NW / (8 * 256), 256, 0, stream>>>(WK, WKb);
    cvt_f32_bf16_kernel<<<NW / (8 * 256), 256, 0, stream>>>(WV, WVb);
    cvt_f32_bf16_kernel<<<NW / (8 * 256), 256, 0, stream>>>(WO, WOb);

    // 1) Q/K/V projections: grid (BS/64, D/256, 3)
    dim3 gproj(BS / 64, DMODEL / 256, 3);
    qkv_proj_kernel<<<gproj, 256, 0, stream>>>(xb, WQb, WKb, WVb, Qb, Kb, Vt);

    // 2) fused causal flash attention: grid (S/64, H, B), 4 waves/block
    dim3 gattn(SEQ / 64, HEADS, BATCH);
    flash_attn_kernel<<<gattn, 128, 0, stream>>>(Qb, Kb, Vt, attnb);

    // 3) output projection: grid (BS/64, D/256)
    dim3 gout(BS / 64, DMODEL / 256, 1);
    out_proj_kernel<<<gout, 256, 0, stream>>>(attnb, WOb, out);
}
